// SimpleLSTM_62285615727291
// MI455X (gfx1250) — compile-verified, matrix operand order checked by reading, not run on hardware
//
#include <hip/hip_runtime.h>
#include <hip/hip_bf16.h>

// ---------------------------------------------------------------------------
// LSTM  (B=64, T=512, F=1024, H=1024), flax OptimizedLSTMCell math.
// Phase 1: xz = x @ Wi + b              (bf16 WMMA GEMM, f32 accumulate)
// Phase 2: ONE persistent scan kernel:  for t: z = xz[:,t] + h@Wh ; gates ;
//          c,h update ; y[:,t] = h, with a grid-wide atomic barrier between
//          steps (64 small blocks -> always co-resident). WhT (8MB bf16)
//          stays hot in the 192MB L2 across all 512 steps.
// ---------------------------------------------------------------------------

#define B_  64
#define T_  512
#define F_  1024
#define H_  1024
#define G4_ 4096   // 4*H
#define NBLK_SCAN 64

typedef __attribute__((ext_vector_type(16))) __bf16 v16bf;
typedef __attribute__((ext_vector_type(8)))  float  v8f;

union V16U { v16bf v; uint4 q[2]; };

// f32 -> bf16 round-to-nearest-even (bit trick)
__device__ __forceinline__ unsigned short f2bf_bits(float f) {
  unsigned u = __float_as_uint(f);
  return (unsigned short)((u + 0x7FFFu + ((u >> 16) & 1u)) >> 16);
}
__device__ __forceinline__ __bf16 f2bf(float f) {
  unsigned short s = f2bf_bits(f);
  __bf16 r;
  __builtin_memcpy(&r, &s, 2);
  return r;
}

__device__ __forceinline__ float fast_sigmoid(float x) {
  return 1.0f / (1.0f + __expf(-x));
}
__device__ __forceinline__ float fast_tanh(float x) {
  float e = __expf(2.0f * x);
  return (e - 1.0f) / (e + 1.0f);
}

// Load one 16x32 bf16 A/B fragment in the CDNA5 WMMA VGPR layout.
// 16-bit A 16x32 (ISA 7.12.2): lanes 0-15 hold row M=lane, K=0..7 in
// VGPR0-3 and K=16..23 in VGPR4-7; lanes 16-31 hold the same rows, K=8..15
// and K=24..31.  B (via pre-transposed N-major weights) uses the identical
// pattern with N as the "row".  Two contiguous 16B loads per lane.
__device__ __forceinline__ v16bf load_ab_tile(const __bf16* __restrict__ base,
                                              int row_stride, int lane) {
  const int half = lane >> 4;
  const int r    = lane & 15;
  const __bf16* p = base + (size_t)r * row_stride + half * 8;
  V16U u;
  u.q[0] = *(const uint4*)(p);
  u.q[1] = *(const uint4*)(p + 16);
  return u.v;
}

// ---------------------------------------------------------------------------
// Pre-passes
// ---------------------------------------------------------------------------

__global__ void __launch_bounds__(256)
k_convert_bf16_x4(const float* __restrict__ src, __bf16* __restrict__ dst,
                  size_t n4) {
  size_t i = (size_t)blockIdx.x * 256 + threadIdx.x;
  if (i >= n4) return;
  float4 f = ((const float4*)src)[i];
  unsigned p0 = (unsigned)f2bf_bits(f.x) | ((unsigned)f2bf_bits(f.y) << 16);
  unsigned p1 = (unsigned)f2bf_bits(f.z) | ((unsigned)f2bf_bits(f.w) << 16);
  ((uint2*)dst)[i] = make_uint2(p0, p1);
}

// W[K][N] f32  ->  WT[N][K] bf16, LDS-tiled 32x32
__global__ void __launch_bounds__(256)
k_transpose_bf16(const float* __restrict__ W, __bf16* __restrict__ WT,
                 int K, int N) {
  __shared__ __bf16 tile[32][33];
  const int kb = blockIdx.x * 32;
  const int nb = blockIdx.y * 32;
  const int tx = threadIdx.x & 31;
  const int ty = threadIdx.x >> 5;
  #pragma unroll
  for (int i = 0; i < 32; i += 8)
    tile[ty + i][tx] = f2bf(W[(size_t)(kb + ty + i) * N + nb + tx]);
  __syncthreads();
  #pragma unroll
  for (int i = 0; i < 32; i += 8)
    WT[(size_t)(nb + ty + i) * K + kb + tx] = tile[tx][ty + i];
}

// c_ws = c0 ; h_bf16 = bf16(h0) ; zero the grid-barrier counter
__global__ void __launch_bounds__(256)
k_init_state(const float* __restrict__ c0, const float* __restrict__ h0,
             float* __restrict__ c_ws, __bf16* __restrict__ h_ws,
             unsigned* __restrict__ sync, int n) {
  int i = blockIdx.x * 256 + threadIdx.x;
  if (i == 0) *sync = 0u;
  if (i >= n) return;
  c_ws[i] = c0[i];
  h_ws[i] = f2bf(h0[i]);
}

// ---------------------------------------------------------------------------
// Phase 1: xz = x @ Wi + b.   One wave per 16(M) x 64(N) strip, K-loop
// software-pipelined (loads for k+32 issued before WMMAs for k).
// ---------------------------------------------------------------------------
__global__ void __launch_bounds__(256)
k_gemm_xz(const __bf16* __restrict__ xbf, const __bf16* __restrict__ WiT,
          const float* __restrict__ bias, float* __restrict__ xz) {
  const int lane   = threadIdx.x & 31;
  const int wv     = (blockIdx.x << 3) + (threadIdx.x >> 5);
  const int n0     = (wv & 63) << 6;          // 64 strips of 64 columns
  const int m0     = (wv >> 6) << 4;          // 16-row tile
  const int half   = lane >> 4;
  const int col    = lane & 15;

  v8f acc[4];
  #pragma unroll
  for (int s = 0; s < 4; ++s) {
    const float bv = bias[n0 + s * 16 + col];
    #pragma unroll
    for (int r = 0; r < 8; ++r) acc[s][r] = bv;
  }

  const __bf16* a_base = xbf + (size_t)m0 * F_;
  v16bf a_cur = load_ab_tile(a_base, F_, lane);
  v16bf b_cur[4];
  #pragma unroll
  for (int s = 0; s < 4; ++s)
    b_cur[s] = load_ab_tile(WiT + (size_t)(n0 + s * 16) * F_, F_, lane);

  #pragma unroll 2
  for (int k = 32; k < F_; k += 32) {
    v16bf a_nxt = load_ab_tile(a_base + k, F_, lane);
    v16bf b_nxt[4];
    #pragma unroll
    for (int s = 0; s < 4; ++s)
      b_nxt[s] = load_ab_tile(WiT + (size_t)(n0 + s * 16) * F_ + k, F_, lane);
    #pragma unroll
    for (int s = 0; s < 4; ++s)
      acc[s] = __builtin_amdgcn_wmma_f32_16x16x32_bf16(
          false, a_cur, false, b_cur[s], (short)0, acc[s], false, false);
    a_cur = a_nxt;
    #pragma unroll
    for (int s = 0; s < 4; ++s) b_cur[s] = b_nxt[s];
  }
  #pragma unroll
  for (int s = 0; s < 4; ++s)
    acc[s] = __builtin_amdgcn_wmma_f32_16x16x32_bf16(
        false, a_cur, false, b_cur[s], (short)0, acc[s], false, false);

  #pragma unroll
  for (int s = 0; s < 4; ++s) {
    const int n = n0 + s * 16 + col;
    #pragma unroll
    for (int r = 0; r < 8; ++r) {
      const int m = m0 + r + half * 8;
      xz[(size_t)m * G4_ + n] = acc[s][r];
    }
  }
}

// ---------------------------------------------------------------------------
// Phase 2: persistent scan kernel.  256 waves (64 blocks x 128 threads).
// Each wave owns 16 batch rows x 16 hidden cols x all 4 gates.  Grid-wide
// barrier between steps via monotonic device-scope atomic counter.
// ---------------------------------------------------------------------------
__global__ void __launch_bounds__(128, 1)
k_lstm_scan(const __bf16* __restrict__ WhT, const float* __restrict__ xz,
            float* __restrict__ cstate, __bf16* __restrict__ hA,
            __bf16* __restrict__ hB, float* __restrict__ y,
            unsigned* __restrict__ sync) {
  const int lane = threadIdx.x & 31;
  const int wv   = (blockIdx.x << 2) + (threadIdx.x >> 5);
  const int jt   = wv & 63;                   // 64 j-tiles of 16 columns
  const int m0   = (wv >> 6) << 4;            // batch tile: 0,16,32,48
  const int j0   = jt << 4;
  const int half = lane >> 4;
  const int col  = lane & 15;
  const int j    = j0 + col;

  // Per-gate B base pointers (fixed for all steps; data stays in L2)
  const __bf16* wh_base[4];
  #pragma unroll
  for (int g = 0; g < 4; ++g)
    wh_base[g] = WhT + (size_t)(g * H_ + j0) * H_;

  for (int t = 0; t < T_; ++t) {
    const __bf16* hin  = (t & 1) ? hB : hA;
    __bf16*       hout = (t & 1) ? hA : hB;

    // acc[g] initialized from xz[:, t, g*H + j]  (i,f,g,o gate order)
    v8f acc[4];
    #pragma unroll
    for (int g = 0; g < 4; ++g) {
      #pragma unroll
      for (int r = 0; r < 8; ++r) {
        const int b = m0 + r + half * 8;
        acc[g][r] = xz[((size_t)b * T_ + t) * G4_ + g * H_ + j];
      }
    }

    // ---- software-pipelined K loop:  z += h @ Wh ----
    const __bf16* a_base = hin + (size_t)m0 * H_;
    v16bf a_cur = load_ab_tile(a_base, H_, lane);
    v16bf b_cur[4];
    #pragma unroll
    for (int g = 0; g < 4; ++g)
      b_cur[g] = load_ab_tile(wh_base[g], H_, lane);

    #pragma unroll 2
    for (int k = 32; k < H_; k += 32) {
      v16bf a_nxt = load_ab_tile(a_base + k, H_, lane);
      v16bf b_nxt[4];
      #pragma unroll
      for (int g = 0; g < 4; ++g)
        b_nxt[g] = load_ab_tile(wh_base[g] + k, H_, lane);
      #pragma unroll
      for (int g = 0; g < 4; ++g)
        acc[g] = __builtin_amdgcn_wmma_f32_16x16x32_bf16(
            false, a_cur, false, b_cur[g], (short)0, acc[g], false, false);
      a_cur = a_nxt;
      #pragma unroll
      for (int g = 0; g < 4; ++g) b_cur[g] = b_nxt[g];
    }
    #pragma unroll
    for (int g = 0; g < 4; ++g)
      acc[g] = __builtin_amdgcn_wmma_f32_16x16x32_bf16(
          false, a_cur, false, b_cur[g], (short)0, acc[g], false, false);

    // ---- gates + state update, entirely in registers ----
    #pragma unroll
    for (int r = 0; r < 8; ++r) {
      const int b = m0 + r + half * 8;
      const float ig = fast_sigmoid(acc[0][r]);
      const float fg = fast_sigmoid(acc[1][r]);
      const float gg = fast_tanh(acc[2][r]);
      const float og = fast_sigmoid(acc[3][r]);
      const size_t idx = (size_t)b * H_ + j;
      const float cn = fg * cstate[idx] + ig * gg;
      const float hn = og * fast_tanh(cn);
      cstate[idx] = cn;
      hout[idx]   = f2bf(hn);
      y[((size_t)b * T_ + t) * H_ + j] = hn;
    }

    // ---- prefetch next step's xz slice (hides HBM latency behind barrier)
    if (t + 1 < T_) {
      #pragma unroll
      for (int g = 0; g < 4; ++g) {
        #pragma unroll
        for (int r = 0; r < 8; ++r) {
          const int b = m0 + r + half * 8;
          __builtin_prefetch(&xz[((size_t)b * T_ + (t + 1)) * G4_ + g * H_ + j],
                             0, 0);
        }
      }
    }

    // ---- grid-wide barrier (release stores, acquire loads) ----
    __threadfence();
    __syncthreads();
    if (threadIdx.x == 0) {
      __hip_atomic_fetch_add(sync, 1u, __ATOMIC_ACQ_REL,
                             __HIP_MEMORY_SCOPE_AGENT);
      const unsigned target = (unsigned)NBLK_SCAN * (unsigned)(t + 1);
      while (__hip_atomic_load(sync, __ATOMIC_ACQUIRE,
                               __HIP_MEMORY_SCOPE_AGENT) < target) {
        __builtin_amdgcn_s_sleep(2);
      }
    }
    __syncthreads();
  }
}

// ---------------------------------------------------------------------------
// Host launcher
// ---------------------------------------------------------------------------
extern "C" void kernel_launch(void* const* d_in, const int* in_sizes, int n_in,
                              void* d_out, int out_size, void* d_ws,
                              size_t ws_size, hipStream_t stream) {
  const float* x  = (const float*)d_in[0];   // [B,T,F]
  const float* c0 = (const float*)d_in[1];   // [B,H]
  const float* h0 = (const float*)d_in[2];   // [B,H]
  const float* Wi = (const float*)d_in[3];   // [F,4H]
  const float* Wh = (const float*)d_in[4];   // [H,4H]
  const float* bb = (const float*)d_in[5];   // [4H]
  float* y = (float*)d_out;                  // [B,T,H]

  uintptr_t p = ((uintptr_t)d_ws + 255) & ~(uintptr_t)255;
  auto carve = [&p](size_t bytes) -> void* {
    void* r = (void*)p;
    p += (bytes + 255) & ~(size_t)255;
    return r;
  };
  __bf16*   xbf  = (__bf16*)carve((size_t)B_ * T_ * F_ * 2);  //  64 MB
  __bf16*   WiT  = (__bf16*)carve((size_t)F_ * G4_ * 2);      //   8 MB
  __bf16*   WhT  = (__bf16*)carve((size_t)H_ * G4_ * 2);      //   8 MB
  float*    xz   = (float*)carve((size_t)B_ * T_ * G4_ * 4);  // 512 MB
  float*    cst  = (float*)carve((size_t)B_ * H_ * 4);
  __bf16*   hA   = (__bf16*)carve((size_t)B_ * H_ * 2);
  __bf16*   hB   = (__bf16*)carve((size_t)B_ * H_ * 2);
  unsigned* syn  = (unsigned*)carve(256);

  // --- pre-passes ---
  {
    const size_t n4 = (size_t)B_ * T_ * F_ / 4;
    k_convert_bf16_x4<<<(unsigned)((n4 + 255) / 256), 256, 0, stream>>>(x, xbf, n4);
  }
  {
    dim3 g(F_ / 32, G4_ / 32);
    k_transpose_bf16<<<g, 256, 0, stream>>>(Wi, WiT, F_, G4_);
  }
  {
    dim3 g(H_ / 32, G4_ / 32);
    k_transpose_bf16<<<g, 256, 0, stream>>>(Wh, WhT, H_, G4_);
  }
  k_init_state<<<(B_ * H_ + 255) / 256, 256, 0, stream>>>(c0, h0, cst, hA, syn,
                                                          B_ * H_);

  // --- phase 1: xz = x@Wi + b ---
  {
    const int nwaves = (B_ * T_ / 16) * (G4_ / 64);         // 131072
    k_gemm_xz<<<nwaves / 8, 256, 0, stream>>>(xbf, WiT, bb, xz);
  }

  // --- phase 2: one persistent kernel runs the whole recurrence ---
  k_lstm_scan<<<NBLK_SCAN, 128, 0, stream>>>(WhT, xz, cst, hA, hB, y, syn);
}
